// unit_gcn_83992380440690
// MI455X (gfx1250) — matrix operand order repaired and checked by reading, not compiled, not run on hardware
//
#include <hip/hip_runtime.h>

#define B_N   16384
#define C_IN  64
#define C_OUT 256
#define VV    25
#define NSTAT 409600.0f   /* B_N * VV */
#define BN_EPS 1e-5f

typedef __attribute__((ext_vector_type(16))) _Float16 v16h;
typedef __attribute__((ext_vector_type(8)))  float    v8f;

union H8  { _Float16 h[8];  uint4 u;    };
union H16 { v16h v;         uint4 u[2]; };
union H32 { _Float16 h[32]; uint4 u[4]; };

#if defined(__HIP_DEVICE_COMPILE__) && __has_builtin(__builtin_amdgcn_global_store_async_from_lds_b128)
#define USE_ASYNC_LDS 1
typedef int v4i_async __attribute__((vector_size(16)));
typedef __attribute__((address_space(1))) v4i_async gv4i_t;   // global 128-bit element
typedef __attribute__((address_space(3))) v4i_async lv4i_t;   // LDS 128-bit element
#endif

__device__ __forceinline__ void wait_dscnt0() {
#if defined(__HIP_DEVICE_COMPILE__)
#if __has_builtin(__builtin_amdgcn_s_wait_dscnt)
  __builtin_amdgcn_s_wait_dscnt(0);
#else
  asm volatile("s_wait_dscnt 0x0" ::: "memory");
#endif
#endif
}
__device__ __forceinline__ void wait_asynccnt0() {
#if defined(__HIP_DEVICE_COMPILE__)
#if __has_builtin(__builtin_amdgcn_s_wait_asynccnt)
  __builtin_amdgcn_s_wait_asynccnt(0);
#else
  asm volatile("s_wait_asynccnt 0x0" ::: "memory");
#endif
#endif
}

__device__ __forceinline__ v8f wmma16(v16h a, v16h b, v8f c) {
  return __builtin_amdgcn_wmma_f32_16x16x32_f16(false, a, false, b, (short)0, c, false, false);
}

// A operand from row-major [M][K] storage (strideH halves, mult of 8).
__device__ __forceinline__ v16h load_a16(const _Float16* p, int m0, int k0, int strideH, int lane) {
  int mr = lane & 15, hi = (lane >> 4) & 1;
  const _Float16* q = p + (size_t)(m0 + mr) * strideH + k0 + hi * 8;
  H16 pk;
  pk.u[0] = *(const uint4*)(q);
  pk.u[1] = *(const uint4*)(q + 16);
  return pk.v;
}

// B operand from transposed [N][K] storage.
__device__ __forceinline__ v16h load_b16(const _Float16* p, int n0, int k0, int strideH, int lane) {
  int nr = lane & 15, hi = (lane >> 4) & 1;
  const _Float16* q = p + (size_t)(n0 + nr) * strideH + k0 + hi * 8;
  H16 pk;
  pk.u[0] = *(const uint4*)(q);
  pk.u[1] = *(const uint4*)(q + 16);
  return pk.v;
}

// ---------------- K0: weight fp32->f16 conversion + stat zeroing ----------------
__global__ __launch_bounds__(256) void k_prep(const float* __restrict__ wa, const float* __restrict__ wb,
                                              const float* __restrict__ wd, const float* __restrict__ wr,
                                              _Float16* __restrict__ Wa, _Float16* __restrict__ Wb,
                                              _Float16* __restrict__ Wd, _Float16* __restrict__ Wr,
                                              float* __restrict__ stats) {
  int idx = blockIdx.x * 256 + threadIdx.x;   // grid = 64 blocks -> 16384 threads
  if (idx < 4096) { Wa[idx] = (_Float16)wa[idx]; Wb[idx] = (_Float16)wb[idx]; }
  Wd[idx] = (_Float16)wd[idx];
  Wr[idx] = (_Float16)wr[idx];
  if (idx < 1536) stats[idx] = 0.0f;
}

// ---------------- K1: adjacency + xa + BN statistics (one wave per item) ----------------
__global__ __launch_bounds__(128) void k_graph(
    const float* __restrict__ x, const float* __restrict__ PA, const float* __restrict__ alpha,
    const float* __restrict__ ba, const float* __restrict__ bb,
    const float* __restrict__ bd, const float* __restrict__ br,
    const _Float16* __restrict__ Wa, const _Float16* __restrict__ Wb,
    const _Float16* __restrict__ Wd, const _Float16* __restrict__ Wr,
    _Float16* __restrict__ xaT_g, float* __restrict__ statY, float* __restrict__ statR)
{
  extern __shared__ char smem[];
  float* sPA = (float*)smem;        // 640 (625 used)
  float* sBa = sPA + 640;           // 64
  float* sBb = sBa + 64;            // 64
  float* sBd = sBb + 64;            // 256
  float* sBr = sBd + 256;           // 256
  float* sSY = sBr + 256;           // 512 (sum,sumsq interleaved)
  float* sSR = sSY + 512;           // 512  -> f32 region = 9216 B
  _Float16* wbuf = (_Float16*)(smem + 9216);
  int w = threadIdx.x >> 5, lane = threadIdx.x & 31;
  _Float16* Xh  = wbuf + w * 9216;  // [64][32]  X row-major (K=v)
  _Float16* XT  = Xh + 2048;        // [32][64]  X transposed (B operand, K=c)
  _Float16* Pt  = XT + 2048;        // [32][64]  P^T ; later reused as xa^T [u][c]
  _Float16* Qt  = Pt + 2048;        // [32][64]  Q^T
  _Float16* A3t = Qt + 2048;        // [32][32]  A3^T (B operand for xa)

  for (int i = threadIdx.x; i < 625; i += blockDim.x) sPA[i] = PA[i];
  for (int i = threadIdx.x; i < 64;  i += blockDim.x) { sBa[i] = ba[i]; sBb[i] = bb[i]; }
  for (int i = threadIdx.x; i < 256; i += blockDim.x) { sBd[i] = bd[i]; sBr[i] = br[i]; }
  for (int i = threadIdx.x; i < 512; i += blockDim.x) { sSY[i] = 0.0f; sSR[i] = 0.0f; }
  __syncthreads();

  float alf = alpha[0];
  int gw = blockIdx.x * 4 + w;
  int nwaves = gridDim.x * 4;       // 4096 waves -> exactly 4 items each

  for (int item = gw; item < B_N; item += nwaves) {
#ifdef USE_ASYNC_LDS
    wait_asynccnt0();               // previous item's async xaT store must be done before Pt reuse
#endif
    if (item + nwaves < B_N)
      __builtin_prefetch(x + (size_t)(item + nwaves) * (C_IN * VV) + lane * 50, 0, 1);

    // ---- stage 1: load X, build Xh [c][v] and XT [v][c] (f16, padded) ----
    for (int rr = 0; rr < 2; ++rr) {
      int c = lane + 32 * rr;
      const float* xr = x + (size_t)item * (C_IN * VV) + c * VV;
      H32 row;
      #pragma unroll
      for (int v = 0; v < VV; ++v) row.h[v] = (_Float16)xr[v];
      #pragma unroll
      for (int v = VV; v < 32; ++v) row.h[v] = (_Float16)0.0f;
      uint4* dst = (uint4*)(Xh + c * 32);
      #pragma unroll
      for (int j = 0; j < 4; ++j) dst[j] = row.u[j];
      #pragma unroll
      for (int v = 0; v < VV; ++v) XT[v * 64 + c] = row.h[v];
    }
    if (lane >= VV) {               // zero padded rows v=25..31 of XT
      uint4 z; z.x = z.y = z.z = z.w = 0u;
      uint4* dst = (uint4*)(XT + lane * 64);
      #pragma unroll
      for (int j = 0; j < 8; ++j) dst[j] = z;
    }
    __builtin_amdgcn_wave_barrier();

    // ---- stage 2: P = Wa@X, Q = Wb@X (store transposed + bias, masked cols) ----
    #pragma unroll
    for (int pq = 0; pq < 2; ++pq) {
      const _Float16* W = pq ? Wb : Wa;
      const float* bias = pq ? sBb : sBa;
      _Float16* DT      = pq ? Qt  : Pt;
      for (int mt = 0; mt < 4; ++mt)
        for (int nt = 0; nt < 2; ++nt) {
          v8f acc = {};
          #pragma unroll
          for (int kt = 0; kt < 2; ++kt)
            acc = wmma16(load_a16(W, mt * 16, kt * 32, 64, lane),
                         load_b16(XT, nt * 16, kt * 32, 64, lane), acc);
          int n  = nt * 16 + (lane & 15);
          int m0 = mt * 16 + (lane >> 4) * 8;
          float keep = (n < VV) ? 1.0f : 0.0f;
          H8 o;
          #pragma unroll
          for (int r = 0; r < 8; ++r)
            o.h[r] = (_Float16)((acc[r] + bias[m0 + r]) * keep);
          *(uint4*)(DT + n * 64 + m0) = o.u;
        }
    }
    __builtin_amdgcn_wave_barrier();

    // ---- stage 3: attn = P^T @ Q ; A3 = PA + alpha*tanh(attn/64) -> A3t [u][v] ----
    for (int mt = 0; mt < 2; ++mt)
      for (int nt = 0; nt < 2; ++nt) {
        v8f acc = {};
        #pragma unroll
        for (int kt = 0; kt < 2; ++kt)
          acc = wmma16(load_a16(Pt, mt * 16, kt * 32, 64, lane),
                       load_b16(Qt, nt * 16, kt * 32, 64, lane), acc);
        int u  = nt * 16 + (lane & 15);
        int m0 = mt * 16 + (lane >> 4) * 8;
        float ku = (u < VV) ? 1.0f : 0.0f;
        int  uc  = (u < VV) ? u : 0;
        H8 o;
        #pragma unroll
        for (int r = 0; r < 8; ++r) {
          int m = m0 + r;
          float km = (m < VV) ? ku : 0.0f;
          int  mc  = (m < VV) ? m : 0;
          float val = (sPA[mc * VV + uc] + alf * tanhf(acc[r] * (1.0f / 64.0f))) * km;
          o.h[r] = (_Float16)val;
        }
        *(uint4*)(A3t + u * 32 + m0) = o.u;
      }
    __builtin_amdgcn_wave_barrier();

    // ---- stage 4: xa = X @ A3, stored transposed [u][c] into Pt, copied to ws ----
    for (int mt = 0; mt < 4; ++mt)
      for (int nt = 0; nt < 2; ++nt) {
        v8f acc = {};
        acc = wmma16(load_a16(Xh, mt * 16, 0, 32, lane),
                     load_b16(A3t, nt * 16, 0, 32, lane), acc);
        int u  = nt * 16 + (lane & 15);
        int c0 = mt * 16 + (lane >> 4) * 8;
        float keep = (u < VV) ? 1.0f : 0.0f;
        H8 o;
        #pragma unroll
        for (int r = 0; r < 8; ++r)
          o.h[r] = (_Float16)(acc[r] * keep);
        *(uint4*)(Pt + u * 64 + c0) = o.u;
      }
    __builtin_amdgcn_wave_barrier();

#ifdef USE_ASYNC_LDS
    {
      wait_dscnt0();   // async engine must observe the completed ds_stores of Pt
      gv4i_t* gbase = (gv4i_t*)(unsigned long long)(xaT_g + (size_t)item * 2048);
      lv4i_t* lbase = (lv4i_t*)(unsigned int)(unsigned long long)Pt;
      #pragma unroll
      for (int i = 0; i < 8; ++i)
        __builtin_amdgcn_global_store_async_from_lds_b128(
            gbase + (lane + i * 32), lbase + (lane + i * 32), 0, 0);
    }
#else
    {
      uint4* d4 = (uint4*)(xaT_g + (size_t)item * 2048);
      const uint4* s4 = (const uint4*)Pt;
      for (int i = lane; i < 256; i += 32) d4[i] = s4[i];
    }
#endif

    // ---- stage 5: per-channel stats of Wd@xa+bd and Wr@X+br (no store) ----
    #pragma unroll
    for (int yr = 0; yr < 2; ++yr) {
      const _Float16* W = yr ? Wr : Wd;
      const float* bias = yr ? sBr : sBd;
      const _Float16* Bm = yr ? XT : Pt;     // XT=[v][c], Pt=xa^T [u][c]
      float* accS = yr ? sSR : sSY;
      for (int mt = 0; mt < 16; ++mt) {
        int m0 = mt * 16 + (lane >> 4) * 8;
        float bv[8];
        #pragma unroll
        for (int r = 0; r < 8; ++r) bv[r] = bias[m0 + r];
        float s1[8] = {}, s2[8] = {};
        for (int nt = 0; nt < 2; ++nt) {
          v8f acc = {};
          #pragma unroll
          for (int kt = 0; kt < 2; ++kt)
            acc = wmma16(load_a16(W, mt * 16, kt * 32, 64, lane),
                         load_b16(Bm, nt * 16, kt * 32, 64, lane), acc);
          int u = nt * 16 + (lane & 15);
          float ku = (u < VV) ? 1.0f : 0.0f;
          #pragma unroll
          for (int r = 0; r < 8; ++r) {
            float val = acc[r] + bv[r];
            s1[r] += val * ku;
            s2[r] += val * val * ku;
          }
        }
        #pragma unroll
        for (int off = 1; off < 16; off <<= 1) {
          #pragma unroll
          for (int r = 0; r < 8; ++r) {
            s1[r] += __shfl_xor(s1[r], off, 32);
            s2[r] += __shfl_xor(s2[r], off, 32);
          }
        }
        if ((lane & 15) == 0) {
          #pragma unroll
          for (int r = 0; r < 8; ++r) {
            atomicAdd(&accS[(m0 + r) * 2 + 0], s1[r]);
            atomicAdd(&accS[(m0 + r) * 2 + 1], s2[r]);
          }
        }
      }
    }
  }
#ifdef USE_ASYNC_LDS
  wait_asynccnt0();
#endif
  __syncthreads();
  for (int i = threadIdx.x; i < 512; i += blockDim.x) {
    atomicAdd(&statY[i], sSY[i]);
    atomicAdd(&statR[i], sSR[i]);
  }
}

// ---------------- K2/K4: fold stats into BN scale/bias ----------------
__global__ void k_bnparams(const float* __restrict__ stat, const float* __restrict__ g,
                           const float* __restrict__ b, float* __restrict__ scb) {
  int c = threadIdx.x;  // 256 threads
  float mean = stat[2 * c] * (1.0f / NSTAT);
  float var  = stat[2 * c + 1] * (1.0f / NSTAT) - mean * mean;
  float sc = g[c] * rsqrtf(var + BN_EPS);
  scb[c] = sc;
  scb[256 + c] = b[c] - mean * sc;
}

// ---------------- K3: recompute y/res, BN+add+ReLU, SE, write y2, gbn stats ----------------
__global__ __launch_bounds__(128) void k_se(
    const float* __restrict__ x, const _Float16* __restrict__ xaT_g,
    const _Float16* __restrict__ Wd, const _Float16* __restrict__ Wr,
    const float* __restrict__ bd, const float* __restrict__ br,
    const float* __restrict__ scbY, const float* __restrict__ scbR,
    const float* __restrict__ fc1w, const float* __restrict__ fc1b,
    const float* __restrict__ fc2w, const float* __restrict__ fc2b,
    float* __restrict__ out, float* __restrict__ statG)
{
  extern __shared__ char smem[];
  float* sSG = (float*)smem;   // 512
  float* sbd = sSG + 512;      // 256
  float* sbr = sbd + 256;      // 256
  float* sY  = sbr + 256;      // 512 scale|bias
  float* sR  = sY + 512;       // 512 -> 8192 B shared
  char* wbp = smem + 8192;
  int w = threadIdx.x >> 5, lane = threadIdx.x & 31;
  char* wbase = wbp + w * 23040;
  _Float16* XT   = (_Float16*)wbase;           // [32][64]
  _Float16* y1t  = (_Float16*)(wbase + 4096);  // [32][256]  y1 transposed [u][m]
  float* fe   = (float*)(wbase + 20480);       // 256
  float* fe1  = (float*)(wbase + 21504);       // 128
  float* sfac = (float*)(wbase + 22016);       // 256

  for (int i = threadIdx.x; i < 512; i += blockDim.x) { sSG[i] = 0.0f; sY[i] = scbY[i]; sR[i] = scbR[i]; }
  for (int i = threadIdx.x; i < 256; i += blockDim.x) { sbd[i] = bd[i]; sbr[i] = br[i]; }
  __syncthreads();

  int gw = blockIdx.x * 4 + w;
  int nw = gridDim.x * 4;
  for (int item = gw; item < B_N; item += nw) {
    if (item + nw < B_N)
      __builtin_prefetch(x + (size_t)(item + nw) * (C_IN * VV) + lane * 50, 0, 1);
    for (int rr = 0; rr < 2; ++rr) {
      int c = lane + 32 * rr;
      const float* xr = x + (size_t)item * (C_IN * VV) + c * VV;
      #pragma unroll
      for (int v = 0; v < VV; ++v) XT[v * 64 + c] = (_Float16)xr[v];
    }
    if (lane >= VV) {
      uint4 z; z.x = z.y = z.z = z.w = 0u;
      uint4* dst = (uint4*)(XT + lane * 64);
      #pragma unroll
      for (int j = 0; j < 8; ++j) dst[j] = z;
    }
    for (int i = lane; i < 256; i += 32) fe[i] = 0.0f;
    __builtin_amdgcn_wave_barrier();
    const _Float16* xaT = xaT_g + (size_t)item * 2048;

    for (int mt = 0; mt < 16; ++mt) {
      int m0 = mt * 16 + (lane >> 4) * 8;
      float cy[8], cr[8], c0v[8];
      #pragma unroll
      for (int r = 0; r < 8; ++r) {
        int m = m0 + r;
        cy[r] = sY[m];
        cr[r] = sR[m];
        c0v[r] = sY[m] * sbd[m] + sY[256 + m] + sR[m] * sbr[m] + sR[256 + m];
      }
      float fs[8] = {};
      for (int nt = 0; nt < 2; ++nt) {
        v8f aY = {}, aR = {};
        #pragma unroll
        for (int kt = 0; kt < 2; ++kt) {
          aY = wmma16(load_a16(Wd, mt * 16, kt * 32, 64, lane),
                      load_b16(xaT, nt * 16, kt * 32, 64, lane), aY);
          aR = wmma16(load_a16(Wr, mt * 16, kt * 32, 64, lane),
                      load_b16(XT,  nt * 16, kt * 32, 64, lane), aR);
        }
        int u = nt * 16 + (lane & 15);
        float ku = (u < VV) ? 1.0f : 0.0f;
        H8 o;
        #pragma unroll
        for (int r = 0; r < 8; ++r) {
          float v = fmaxf(cy[r] * aY[r] + cr[r] * aR[r] + c0v[r], 0.0f);
          fs[r] += v * ku;
          o.h[r] = (_Float16)v;
        }
        *(uint4*)(y1t + u * 256 + m0) = o.u;   // contiguous transposed store
      }
      #pragma unroll
      for (int off = 1; off < 16; off <<= 1) {
        #pragma unroll
        for (int r = 0; r < 8; ++r) fs[r] += __shfl_xor(fs[r], off, 32);
      }
      if ((lane & 15) == 0) {
        #pragma unroll
        for (int r = 0; r < 8; ++r) atomicAdd(&fe[m0 + r], fs[r]);
      }
    }
    __builtin_amdgcn_wave_barrier();

    // SE MLP (wave-local; fe holds SUM over u, mean = /25)
    #pragma unroll
    for (int j = 0; j < 4; ++j) {
      int o = lane * 4 + j;
      const float* wrow = fc1w + o * 256;
      float dot = 0.0f;
      for (int c2 = 0; c2 < 256; ++c2) dot += wrow[c2] * fe[c2];
      fe1[o] = fmaxf(fc1b[o] + dot * (1.0f / (float)VV), 0.0f);
    }
    __builtin_amdgcn_wave_barrier();
    #pragma unroll
    for (int j = 0; j < 8; ++j) {
      int o = lane * 8 + j;
      const float* wrow = fc2w + o * 128;
      float dot = 0.0f;
      for (int c2 = 0; c2 < 128; ++c2) dot += wrow[c2] * fe1[c2];
      float s = fc2b[o] + dot;
      sfac[o] = 1.0f + 1.0f / (1.0f + __expf(-s));
    }
    __builtin_amdgcn_wave_barrier();

    // y2 = y1 * (1 + sigmoid), write out, accumulate gbn stats
    for (int rr = 0; rr < 8; ++rr) {
      int m = lane + 32 * rr;
      float sc = sfac[m];
      float s1 = 0.0f, s2 = 0.0f;
      float* orow = out + (size_t)item * (C_OUT * VV) + m * VV;
      #pragma unroll
      for (int u = 0; u < VV; ++u) {
        float v = (float)y1t[u * 256 + m] * sc;
        orow[u] = v;
        s1 += v; s2 += v * v;
      }
      atomicAdd(&sSG[m * 2 + 0], s1);
      atomicAdd(&sSG[m * 2 + 1], s2);
    }
  }
  __syncthreads();
  for (int i = threadIdx.x; i < 512; i += blockDim.x) atomicAdd(&statG[i], sSG[i]);
}

// ---------------- K5: in-place final BN + ReLU ----------------
__global__ __launch_bounds__(256) void k_final(float* __restrict__ out, const float* __restrict__ scbG) {
  const size_t total = (size_t)B_N * C_OUT * VV;
  for (size_t i = (size_t)blockIdx.x * blockDim.x + threadIdx.x; i < total;
       i += (size_t)gridDim.x * blockDim.x) {
    int m = (int)((i / VV) % C_OUT);
    float v = out[i] * scbG[m] + scbG[256 + m];
    out[i] = fmaxf(v, 0.0f);
  }
}

extern "C" void kernel_launch(void* const* d_in, const int* in_sizes, int n_in,
                              void* d_out, int out_size, void* d_ws, size_t ws_size,
                              hipStream_t stream) {
  const float* x    = (const float*)d_in[0];
  const float* PA   = (const float*)d_in[1];
  const float* alpha= (const float*)d_in[2];
  const float* wa   = (const float*)d_in[3];
  const float* ba   = (const float*)d_in[4];
  const float* wbw  = (const float*)d_in[5];
  const float* bb   = (const float*)d_in[6];
  const float* wd   = (const float*)d_in[7];
  const float* bd   = (const float*)d_in[8];
  const float* bng  = (const float*)d_in[9];
  const float* bnb  = (const float*)d_in[10];
  const float* wr   = (const float*)d_in[11];
  const float* brr  = (const float*)d_in[12];
  const float* rbg  = (const float*)d_in[13];
  const float* rbb  = (const float*)d_in[14];
  const float* fc1w = (const float*)d_in[15];
  const float* fc1b = (const float*)d_in[16];
  const float* fc2w = (const float*)d_in[17];
  const float* fc2b = (const float*)d_in[18];
  const float* gg   = (const float*)d_in[19];
  const float* gb   = (const float*)d_in[20];
  float* out = (float*)d_out;
  char* ws = (char*)d_ws;

  _Float16* Wa_h = (_Float16*)(ws + 0);
  _Float16* Wb_h = (_Float16*)(ws + 8192);
  _Float16* Wd_h = (_Float16*)(ws + 16384);
  _Float16* Wr_h = (_Float16*)(ws + 49152);
  float* stats   = (float*)(ws + 81920);          // statY|statR|statG, 512 f32 each
  float* statY = stats, *statR = stats + 512, *statG = stats + 1024;
  float* scbY = (float*)(ws + 88064);
  float* scbR = (float*)(ws + 90112);
  float* scbG = (float*)(ws + 92160);
  _Float16* xaT = (_Float16*)(ws + 98304);        // [B][32][64] f16 = 64 MiB

  k_prep<<<64, 256, 0, stream>>>(wa, wbw, wd, wr, Wa_h, Wb_h, Wd_h, Wr_h, stats);
  k_graph<<<1024, 128, 82944, stream>>>(x, PA, alpha, ba, bb, bd, brr,
                                        Wa_h, Wb_h, Wd_h, Wr_h, xaT, statY, statR);
  k_bnparams<<<1, 256, 0, stream>>>(statY, bng, bnb, scbY);
  k_bnparams<<<1, 256, 0, stream>>>(statR, rbg, rbb, scbR);
  k_se<<<1024, 128, 100352, stream>>>(x, xaT, Wd_h, Wr_h, bd, brr, scbY, scbR,
                                      fc1w, fc1b, fc2w, fc2b, out, statG);
  k_bnparams<<<1, 256, 0, stream>>>(statG, gg, gb, scbG);
  k_final<<<4096, 256, 0, stream>>>(out, scbG);
  (void)in_sizes; (void)n_in; (void)out_size; (void)ws_size;
}
